// SO2EquivariantGraphAttention_52836687675590
// MI455X (gfx1250) — matrix operand order repaired and checked by reading, not compiled, not run on hardware
//
#include <hip/hip_runtime.h>

// ---------------------------------------------------------------------------
// SO2 equivariant graph attention for gfx1250 (MI455X), wave32 + WMMA bf16.
// All dense stages run through v_wmma_f32_16x16x32_bf16; fp32 accumulation.
// Operands are staged in fragment-order (A) / column-major (B) so every WMMA
// fragment is built from two contiguous 16-byte loads (ds/global _b128).
// The main GEMM DMAs its A tiles with GLOBAL_LOAD_ASYNC_TO_LDS_B128 into a
// double-buffered LDS tile, overlapping the next tile's DMA with the WMMAs.
// ---------------------------------------------------------------------------

typedef __attribute__((ext_vector_type(16))) __bf16 v16bf;
typedef __attribute__((ext_vector_type(8)))  __bf16 v8bf;
typedef __attribute__((ext_vector_type(8)))  float  v8f;

__constant__ int d_PERM[25]    = {0,2,6,12,20, 1,5,11,19, 3,7,13,21, 4,10,18, 8,14,22, 9,17, 15,23, 16,24};
__constant__ int d_INVPERM[25] = {0,5,1,9,13, 6,2,10,16,19, 14,7,3,11,17, 21,23,20,15,8, 4,12,18,22,24};
__constant__ int d_CUM[4]      = {0,1024,1792,2304};

__device__ __forceinline__ __bf16 to_bf16(float f) {
  unsigned u = __float_as_uint(f);
  unsigned r = u + 0x7FFFu + ((u >> 16) & 1u);   // round-to-nearest-even
  unsigned short h = (unsigned short)(r >> 16);
  __bf16 b;
  __builtin_memcpy(&b, &h, 2);
  return b;
}

__device__ __forceinline__ float siluf(float v) { return v / (1.f + expf(-v)); }

__device__ __forceinline__ void block_of(int p, int& bi, int& off, int& sz) {
  if (p < 13)      { bi = 0; off = 5;  sz = 4; }
  else if (p < 19) { bi = 1; off = 13; sz = 3; }
  else if (p < 23) { bi = 2; off = 19; sz = 2; }
  else             { bi = 3; off = 23; sz = 1; }
}

// A-operand slot within a 32-wide K block: contiguous order, matching B
// (k0-15 -> slots 0-15, k16-31 -> slots 16-31).
__device__ __forceinline__ int a_slot(int k) {
  int hi = (k >> 4) & 1, kp = k & 15;
  return hi * 16 + kp;
}

// Two contiguous 16B loads -> one 16-element bf16 fragment.
__device__ __forceinline__ v16bf load16(const __bf16* p) {
  v8bf lo = *(const v8bf*)p;
  v8bf hi = *(const v8bf*)(p + 8);
  return __builtin_shufflevector(lo, hi, 0,1,2,3,4,5,6,7,8,9,10,11,12,13,14,15);
}

__device__ __forceinline__ v8f wmma_bf16(v16bf a, v16bf b, v8f c) {
  return __builtin_amdgcn_wmma_f32_16x16x32_bf16(false, a, false, b, (short)0, c, false, false);
}

// ---------------------------------------------------------------------------
// Generic GEMM: C[M,N] = A[M,K] @ B[K,N] (+bias).  BT is B stored column-major
// ([N][K] bf16) so B fragments load straight from global (L2-resident weights).
// A rows use grouped addressing: off(r) = base + (r/R)*GS + (r%R)*RS.
// A tiles are DMA'd via GLOBAL_LOAD_ASYNC_TO_LDS_B128 into double-buffered LDS
// (ASYNCcnt tracked); OOB rows read mapped scratch garbage, which is safe
// because A row r only feeds C row r and the C store is guarded.
// Block = 8 wave32s; tile 128(M) x 64(N); K step 32 -> 4 WMMAs / wave / step.
// K must be a multiple of 32 (true for every stage here).
// ---------------------------------------------------------------------------
__global__ void __launch_bounds__(256)
gemm_bf16(const __bf16* __restrict__ A, const __bf16* __restrict__ BT,
          const float* __restrict__ bias, float* __restrict__ C,
          int M, int N, int K,
          int aR, long aGS, long aRS, long aBase,
          int cR, long cGS, long cRS, long cBase)
{
  __shared__ alignas(16) __bf16 sA[2][128 * 32];   // fragment-order A tiles
  const int m0 = blockIdx.y * 128;
  const int n0 = blockIdx.x * 64;
  const int w  = threadIdx.x >> 5;
  const int lane = threadIdx.x & 31;
  const int hv = lane >> 4, ln = lane & 15;
  v8f acc[4] = {};

  // Issue async DMA of one 128x32 A tile (512 x 16B) into LDS buffer `buf`.
  auto issueA = [&](int buf, int k0) {
#pragma unroll
    for (int u = 0; u < 2; ++u) {
      int t = threadIdx.x + u * 256;
      int rr = t >> 2, c8 = t & 3;
      int row = m0 + rr;
      unsigned dst = (unsigned)(size_t)&sA[buf][rr * 32 + c8 * 8];
      const __bf16* src = A + aBase + (long)(row / aR) * aGS +
                          (long)(row % aR) * aRS + k0 + c8 * 8;
      asm volatile("global_load_async_to_lds_b128 %0, %1, off"
                   :: "v"(dst), "v"(src) : "memory");
    }
  };

  const int nk = K >> 5;
  issueA(0, 0);
  for (int ki = 0; ki < nk; ++ki) {
    asm volatile("s_wait_asynccnt 0x0" ::: "memory");
    __syncthreads();
    if (ki + 1 < nk) issueA((ki + 1) & 1, (ki + 1) << 5);
    const int k0 = ki << 5;
    if (k0 + 32 < K)   // warm next weight columns into GL2 (global_prefetch_b8)
      __builtin_prefetch(BT + (long)(n0 + (threadIdx.x & 63)) * K + k0 + 32, 0, 1);
    const __bf16* base = sA[ki & 1];
    v16bf fa  = load16(&base[(w * 16 + ln) * 32 + hv * 16]);
    v16bf fb0 = load16(BT + (long)(n0 +  0 + ln) * K + k0 + hv * 16);
    v16bf fb1 = load16(BT + (long)(n0 + 16 + ln) * K + k0 + hv * 16);
    v16bf fb2 = load16(BT + (long)(n0 + 32 + ln) * K + k0 + hv * 16);
    v16bf fb3 = load16(BT + (long)(n0 + 48 + ln) * K + k0 + hv * 16);
    acc[0] = wmma_bf16(fa, fb0, acc[0]);
    acc[1] = wmma_bf16(fa, fb1, acc[1]);
    acc[2] = wmma_bf16(fa, fb2, acc[2]);
    acc[3] = wmma_bf16(fa, fb3, acc[3]);
  }
#pragma unroll
  for (int j = 0; j < 4; ++j) {
#pragma unroll
    for (int i = 0; i < 8; ++i) {
      int row = m0 + w * 16 + i + 8 * hv;
      int col = n0 + j * 16 + ln;
      if (row < M && col < N) {
        float v = acc[j][i];
        if (bias) v += bias[col];
        C[cBase + (long)(row / cR) * cGS + (long)(row % cR) * cRS + col] = v;
      }
    }
  }
}

// ---------------------------------------------------------------------------
// Per-edge Wigner rotate fwd: msg_m[e, INVPERM[m], :] = (W[e] @ [x_src|x_tgt])[m,:]
// W padded 25->32 (one K step).  1 block / edge, 8 waves x 4 WMMAs.
// ---------------------------------------------------------------------------
__global__ void __launch_bounds__(256)
wigner_fwd(const float* __restrict__ wig, const float* __restrict__ x,
           const int* __restrict__ ei, float* __restrict__ msg_m, int E)
{
  const int e = blockIdx.x;
  const int src = ei[e], tgt = ei[E + e];
  __shared__ alignas(16) __bf16 sW[32 * 32];    // A, fragment order
  __shared__ alignas(16) __bf16 sX[256 * 32];   // B, column-major [c][k]
  for (int t = threadIdx.x; t < 32 * 32; t += 256) {
    int m = t >> 5, l = t & 31;
    float v = (m < 25 && l < 25) ? wig[(long)e * 625 + m * 25 + l] : 0.f;
    sW[m * 32 + a_slot(l)] = to_bf16(v);
  }
  for (int t = threadIdx.x; t < 32 * 256; t += 256) {
    int c = t & 255, l = t >> 8;
    float v = 0.f;
    if (l < 25) v = (c < 128) ? x[((long)src * 25 + l) * 128 + c]
                              : x[((long)tgt * 25 + l) * 128 + (c - 128)];
    sX[c * 32 + l] = to_bf16(v);
  }
  __syncthreads();
  const int w = threadIdx.x >> 5;
  const int lane = threadIdx.x & 31;
  const int hv = lane >> 4, ln = lane & 15;
#pragma unroll
  for (int q = 0; q < 4; ++q) {
    int f = w * 4 + q;
    int mt = f >> 4, nt = f & 15;
    v16bf fa = load16(&sW[(mt * 16 + ln) * 32 + hv * 16]);
    v16bf fb = load16(&sX[(nt * 16 + ln) * 32 + hv * 16]);
    v8f c = {};
    c = wmma_bf16(fa, fb, c);
#pragma unroll
    for (int i = 0; i < 8; ++i) {
      int row = mt * 16 + i + 8 * hv;
      if (row < 25)
        msg_m[((long)e * 25 + d_INVPERM[row]) * 256 + nt * 16 + ln] = c[i];
    }
  }
}

// ---------------------------------------------------------------------------
// Per-edge inverse Wigner fused with attention weight + scatter-add to nodes:
//   node[tgt[e], l, h*16+c] += alpha[e,h] * (W[e]^T @ msg_l[e])[l, h*16+c]
// ---------------------------------------------------------------------------
__global__ void __launch_bounds__(256)
wigner_bwd_attn(const float* __restrict__ wig, const float* __restrict__ msg2m,
                const float* __restrict__ alpha, const int* __restrict__ ei,
                float* __restrict__ node, int E)
{
  const int e = blockIdx.x;
  const int tgt = ei[E + e];
  __shared__ alignas(16) __bf16 sW[32 * 32];    // W^T, fragment order (k = m)
  __shared__ alignas(16) __bf16 sM[128 * 32];   // msg_l column-major [c][k]
  for (int t = threadIdx.x; t < 32 * 32; t += 256) {
    int l = t >> 5, m = t & 31;
    float v = (l < 25 && m < 25) ? wig[(long)e * 625 + m * 25 + l] : 0.f;
    sW[l * 32 + a_slot(m)] = to_bf16(v);
  }
  for (int t = threadIdx.x; t < 32 * 128; t += 256) {
    int c = t & 127, k = t >> 7;
    float v = (k < 25) ? msg2m[((long)e * 25 + d_INVPERM[k]) * 128 + c] : 0.f;
    sM[c * 32 + k] = to_bf16(v);
  }
  __syncthreads();
  const int w = threadIdx.x >> 5;
  const int lane = threadIdx.x & 31;
  const int hv = lane >> 4, ln = lane & 15;
#pragma unroll
  for (int q = 0; q < 2; ++q) {
    int f = w * 2 + q;
    int mt = f >> 3, nt = f & 7;
    v16bf fa = load16(&sW[(mt * 16 + ln) * 32 + hv * 16]);
    v16bf fb = load16(&sM[(nt * 16 + ln) * 32 + hv * 16]);
    v8f c = {};
    c = wmma_bf16(fa, fb, c);
#pragma unroll
    for (int i = 0; i < 8; ++i) {
      int row = mt * 16 + i + 8 * hv;
      int col = nt * 16 + ln;
      if (row < 25) {
        float a = alpha[(long)e * 8 + (col >> 4)];
        atomicAdd(&node[((long)tgt * 25 + row) * 128 + col], c[i] * a);
      }
    }
  }
}

// ---------------------------------------------------------------------------
// grid = silu(to_grid[100,25] @ msg_l[25,64]) per edge.  A comes pre-swizzled
// from global; result written TRANSPOSED [e][col][128] (k tail zeroed) so the
// second einsum can stream it with contiguous 16B loads.
// ---------------------------------------------------------------------------
__global__ void __launch_bounds__(256)
grid_fwd(const __bf16* __restrict__ tgsw, const float* __restrict__ msg1,
         __bf16* __restrict__ gridT, int E)
{
  const int e = blockIdx.x;
  __shared__ alignas(16) __bf16 sB[64 * 32];    // msg_l column-major [c][k]
  for (int t = threadIdx.x; t < 64 * 32; t += 256) {
    int c = t & 63, k = t >> 6;
    float v = (k < 25) ? msg1[((long)e * 25 + d_INVPERM[k]) * 64 + c] : 0.f;
    sB[c * 32 + k] = to_bf16(v);
  }
  for (int t = threadIdx.x; t < 64 * 16; t += 256) {     // zero k = 112..127
    int col = t >> 4, kz = 112 + (t & 15);
    gridT[((long)e * 64 + col) * 128 + kz] = to_bf16(0.f);
  }
  __syncthreads();
  const int w = threadIdx.x >> 5;
  const int lane = threadIdx.x & 31;
  const int hv = lane >> 4, ln = lane & 15;
  for (int f = w; f < 28; f += 8) {
    int mt = f >> 2, nt = f & 3;
    v16bf fa = load16(tgsw + (mt * 16 + ln) * 32 + hv * 16);
    v16bf fb = load16(&sB[(nt * 16 + ln) * 32 + hv * 16]);
    v8f c = {};
    c = wmma_bf16(fa, fb, c);
#pragma unroll
    for (int i = 0; i < 8; ++i) {
      int row = mt * 16 + i + 8 * hv;       // k index on grid
      int col = nt * 16 + ln;
      gridT[((long)e * 64 + col) * 128 + row] = to_bf16(siluf(c[i]));
    }
  }
}

// s2[25,64] = from_grid^T[25,100] @ grid[100,64] per edge (K padded to 128).
// No LDS: A pre-swizzled global, B transposed global; all 16B loads.
__global__ void __launch_bounds__(256)
grid_bwd(const __bf16* __restrict__ fgsw, const __bf16* __restrict__ gridT,
         float* __restrict__ s2, int E)
{
  const int e = blockIdx.x;
  const int w = threadIdx.x >> 5;
  const int mt = w >> 2, nt = w & 3;
  const int lane = threadIdx.x & 31;
  const int hv = lane >> 4, ln = lane & 15;
  v8f acc = {};
  for (int k0 = 0; k0 < 128; k0 += 32) {
    v16bf fa = load16(fgsw + (mt * 16 + ln) * 128 + k0 + hv * 16);
    v16bf fb = load16(gridT + ((long)e * 64 + nt * 16 + ln) * 128 + k0 + hv * 16);
    acc = wmma_bf16(fa, fb, acc);
  }
#pragma unroll
  for (int i = 0; i < 8; ++i) {
    int row = mt * 16 + i + 8 * hv;
    if (row < 25) s2[((long)e * 25 + row) * 64 + nt * 16 + ln] = acc[i];
  }
}

// ---------------------------- elementwise kernels --------------------------
// out[r][c] = in[r][c] (trans=0) or in[c][r] (trans=1), zero padded, bf16.
__global__ void __launch_bounds__(256)
convert_pad(const float* __restrict__ in, __bf16* __restrict__ out,
            int Rin, int Cin, int Rout, int Cout, int trans)
{
  long i = (long)blockIdx.x * 256 + threadIdx.x;
  if (i >= (long)Rout * Cout) return;
  int r = (int)(i / Cout), c = (int)(i % Cout);
  float v = 0.f;
  if (!trans) { if (r < Rin && c < Cin) v = in[(long)r * Cin + c]; }
  else        { if (r < Cin && c < Rin) v = in[(long)c * Cin + r]; }
  out[i] = to_bf16(v);
}

// A-operand pre-swizzle into [Rout][Kpad] fragment-order blocks (Kpad % 32 == 0)
__global__ void __launch_bounds__(256)
convert_aswz(const float* __restrict__ in, __bf16* __restrict__ out,
             int Rin, int Cin, int Rout, int Kpad, int trans)
{
  long i = (long)blockIdx.x * 256 + threadIdx.x;
  if (i >= (long)Rout * Kpad) return;
  int r = (int)(i / Kpad), k = (int)(i % Kpad);
  float v = 0.f;
  if (!trans) { if (r < Rin && k < Cin) v = in[(long)r * Cin + k]; }
  else        { if (r < Cin && k < Rin) v = in[(long)k * Cin + r]; }
  out[(long)r * Kpad + (k & ~31) + a_slot(k & 31)] = to_bf16(v);
}

__global__ void __launch_bounds__(256)
xedge_kernel(const float* __restrict__ ed, const float* __restrict__ st,
             const float* __restrict__ tt, const int* __restrict__ an,
             const int* __restrict__ ei, __bf16* __restrict__ xe, int E)
{
  long i = (long)blockIdx.x * 256 + threadIdx.x;
  if (i >= (long)E * 256) return;
  int c = (int)(i & 255);
  long e = i >> 8;
  float v;
  if (c < 128)      v = ed[e * 128 + c];
  else if (c < 192) v = st[(long)an[ei[e]] * 64 + (c - 128)];
  else              v = tt[(long)an[ei[E + e]] * 64 + (c - 192)];
  xe[i] = to_bf16(v);
}

__global__ void __launch_bounds__(256)
ln_silu_kernel(const float* __restrict__ in, const float* __restrict__ g,
               const float* __restrict__ b, __bf16* __restrict__ out, int rows)
{
  int r = blockIdx.x * 256 + threadIdx.x;
  if (r >= rows) return;
  const float* xr = in + (long)r * 64;
  float s = 0.f, sq = 0.f;
  for (int i = 0; i < 64; ++i) { float v = xr[i]; s += v; sq += v * v; }
  float mu = s * (1.f / 64.f);
  float var = sq * (1.f / 64.f) - mu * mu;
  float inv = rsqrtf(var + 1e-5f);
  for (int i = 0; i < 64; ++i) {
    float v = (xr[i] - mu) * inv * g[i] + b[i];
    out[(long)r * 64 + i] = to_bf16(siluf(v));
  }
}

__global__ void __launch_bounds__(256)
rad_scale(const float* __restrict__ msg_m, const float* __restrict__ rad,
          __bf16* __restrict__ out, long total)
{
  long i = (long)blockIdx.x * 256 + threadIdx.x;
  if (i >= total) return;
  int c = (int)(i & 255);
  long t = i >> 8;
  int p = (int)(t % 25);
  long e = t / 25;
  int ro;
  if (p < 5) ro = p * 256 + c;
  else {
    int bi, off, sz; block_of(p, bi, off, sz);
    int s = (p - off) % sz;
    ro = 1280 + d_CUM[bi] + s * 256 + c;
  }
  out[i] = to_bf16(msg_m[i] * rad[e * 3840 + ro]);
}

__global__ void __launch_bounds__(256)
combine_kernel(const float* __restrict__ fc0o, int fstride, int fbase,
               const float* __restrict__ ya, const float* __restrict__ yb,
               const float* __restrict__ yc, const float* __restrict__ yd,
               float* __restrict__ outm, int mout, long total)
{
  long i = (long)blockIdx.x * 256 + threadIdx.x;
  if (i >= total) return;
  int c = (int)(i % mout);
  long t = i / mout;
  int p = (int)(t % 25);
  long e = t / 25;
  float v;
  if (p < 5) {
    v = fc0o[e * (long)fstride + fbase + p * mout + c];
  } else {
    int bi, off, sz; block_of(p, bi, off, sz);
    const float* y = bi == 0 ? ya : bi == 1 ? yb : bi == 2 ? yc : yd;
    int Nb = 2 * mout * sz, halfo = mout * sz;
    int d = p - off;
    int j2 = d >= sz ? 1 : 0;
    int s = d - j2 * sz;
    long r0 = (2 * e) * (long)Nb, r1 = (2 * e + 1) * (long)Nb;
    v = (j2 == 0) ? (y[r0 + s * mout + c] - y[r1 + halfo + s * mout + c])
                  : (y[r1 + s * mout + c] + y[r0 + halfo + s * mout + c]);
  }
  outm[i] = v;
}

__global__ void __launch_bounds__(256)
scaled2_kernel(const float* __restrict__ fc0o, const float* __restrict__ s2,
               __bf16* __restrict__ out, long total)
{
  long i = (long)blockIdx.x * 256 + threadIdx.x;
  if (i >= total) return;
  int c = (int)(i & 63);
  long t = i >> 6;
  int p = (int)(t % 25);
  long e = t / 25;
  int q = d_PERM[p];
  float v = (q == 0) ? siluf(fc0o[e * 896 + 512 + c]) : s2[(e * 25 + q) * 64 + c];
  out[i] = to_bf16(v);
}

__global__ void __launch_bounds__(256)
alpha_logits_kernel(const float* __restrict__ fc0o, const float* __restrict__ g,
                    const float* __restrict__ b, const float* __restrict__ dot,
                    float* __restrict__ logits, long EH)
{
  long i = (long)blockIdx.x * 256 + threadIdx.x;
  if (i >= EH) return;
  int h = (int)(i & 7);
  long e = i >> 3;
  const float* xr = fc0o + e * 896 + h * 64;
  float s = 0.f, sq = 0.f;
  for (int c = 0; c < 64; ++c) { float v = xr[c]; s += v; sq += v * v; }
  float mu = s * (1.f / 64.f), var = sq * (1.f / 64.f) - mu * mu;
  float inv = rsqrtf(var + 1e-5f);
  float acc = 0.f;
  for (int c = 0; c < 64; ++c) {
    float a = (xr[c] - mu) * inv * g[c] + b[c];
    float sg = 1.f / (1.f + expf(-a));
    float a2 = 0.6f * a + 0.4f * a * (2.f * sg - 1.f);
    acc += a2 * dot[h * 64 + c];
  }
  logits[i] = acc;
}

__global__ void __launch_bounds__(256)
init_f(float* __restrict__ p, float v, long n)
{
  long i = (long)blockIdx.x * 256 + threadIdx.x;
  if (i < n) p[i] = v;
}

__device__ __forceinline__ void atomicMaxF(float* addr, float val) {
  unsigned* ua = (unsigned*)addr;
  unsigned old = *ua;
  while (__uint_as_float(old) < val) {
    unsigned assumed = old;
    old = atomicCAS(ua, assumed, __float_as_uint(val));
    if (old == assumed) break;
  }
}

__global__ void __launch_bounds__(256)
smax_max(const float* __restrict__ logits, const int* __restrict__ ei,
         float* __restrict__ mx, int E)
{
  long i = (long)blockIdx.x * 256 + threadIdx.x;
  if (i >= (long)E * 8) return;
  int h = (int)(i & 7);
  long e = i >> 3;
  atomicMaxF(&mx[(long)ei[E + e] * 8 + h], logits[i]);
}

__global__ void __launch_bounds__(256)
smax_sum(const float* __restrict__ logits, const int* __restrict__ ei,
         const float* __restrict__ mx, float* __restrict__ ex,
         float* __restrict__ ssum, int E)
{
  long i = (long)blockIdx.x * 256 + threadIdx.x;
  if (i >= (long)E * 8) return;
  int h = (int)(i & 7);
  long e = i >> 3;
  long t = ei[E + e];
  float v = expf(logits[i] - mx[t * 8 + h]);
  ex[i] = v;
  atomicAdd(&ssum[t * 8 + h], v);
}

__global__ void __launch_bounds__(256)
smax_norm(const int* __restrict__ ei, const float* __restrict__ ssum,
          float* __restrict__ ex, int E)
{
  long i = (long)blockIdx.x * 256 + threadIdx.x;
  if (i >= (long)E * 8) return;
  int h = (int)(i & 7);
  long e = i >> 3;
  long t = ei[E + e];
  ex[i] = ex[i] / (ssum[t * 8 + h] + 1e-16f);
}

__global__ void __launch_bounds__(256)
bias_row0(float* __restrict__ out, const float* __restrict__ pb, int NN)
{
  long i = (long)blockIdx.x * 256 + threadIdx.x;
  if (i >= (long)NN * 128) return;
  int c = (int)(i & 127);
  long n = i >> 7;
  out[n * 3200 + c] += pb[c];
}

// ---------------------------------------------------------------------------
extern "C" void kernel_launch(void* const* d_in, const int* in_sizes, int n_in,
                              void* d_out, int out_size, void* d_ws, size_t ws_size,
                              hipStream_t stream)
{
  (void)n_in; (void)out_size; (void)ws_size;
  const float* x     = (const float*)d_in[0];
  const float* edist = (const float*)d_in[1];
  const float* stab  = (const float*)d_in[2];
  const float* ttab  = (const float*)d_in[3];
  const float* rw1   = (const float*)d_in[4];
  const float* rb1   = (const float*)d_in[5];
  const float* rg1   = (const float*)d_in[6];
  const float* rbe1  = (const float*)d_in[7];
  const float* rw2   = (const float*)d_in[8];
  const float* rb2   = (const float*)d_in[9];
  const float* rg2   = (const float*)d_in[10];
  const float* rbe2  = (const float*)d_in[11];
  const float* rw3   = (const float*)d_in[12];
  const float* rb3   = (const float*)d_in[13];
  const float* wig   = (const float*)d_in[14];
  const float* c1fw  = (const float*)d_in[15];
  const float* c1fb  = (const float*)d_in[16];
  const float* c1wm[4] = {(const float*)d_in[17], (const float*)d_in[18],
                          (const float*)d_in[19], (const float*)d_in[20]};
  const float* tgw   = (const float*)d_in[21];
  const float* fgw   = (const float*)d_in[22];
  const float* alg   = (const float*)d_in[23];
  const float* albp  = (const float*)d_in[24];
  const float* adot  = (const float*)d_in[25];
  const float* c2fw  = (const float*)d_in[26];
  const float* c2fb  = (const float*)d_in[27];
  const float* c2wm[4] = {(const float*)d_in[28], (const float*)d_in[29],
                          (const float*)d_in[30], (const float*)d_in[31]};
  const float* pw    = (const float*)d_in[32];
  const float* pb    = (const float*)d_in[33];
  const int*   an    = (const int*)d_in[34];
  const int*   ei    = (const int*)d_in[35];
  float* out = (float*)d_out;

  const int E  = in_sizes[1] / 128;
  const int NN = in_sizes[0] / (25 * 128);
  const int SZ[4]  = {4, 3, 2, 1};
  const int OFF[4] = {5, 13, 19, 23};

  char* cur = (char*)d_ws;
  auto alloc = [&](size_t b) -> void* { void* p = cur; cur += (b + 255) & ~(size_t)255; return p; };

  // bf16 weights: B operands column-major [N][K]; A operands fragment-swizzled
  __bf16* wb_rw1 = (__bf16*)alloc((size_t)256 * 64 * 2);
  __bf16* wb_rw2 = (__bf16*)alloc((size_t)64 * 64 * 2);
  __bf16* wb_rw3 = (__bf16*)alloc((size_t)64 * 3840 * 2);
  __bf16* wb_c1f = (__bf16*)alloc((size_t)1280 * 896 * 2);
  __bf16* wb_c1w[4];
  for (int i = 0; i < 4; ++i) wb_c1w[i] = (__bf16*)alloc((size_t)(SZ[i] * 256) * (2 * 64 * SZ[i]) * 2);
  __bf16* wb_c2f = (__bf16*)alloc((size_t)320 * 640 * 2);
  __bf16* wb_c2w[4];
  for (int i = 0; i < 4; ++i) wb_c2w[i] = (__bf16*)alloc((size_t)(SZ[i] * 64) * (2 * 128 * SZ[i]) * 2);
  __bf16* wb_tg  = (__bf16*)alloc((size_t)112 * 32 * 2);
  __bf16* wb_fgT = (__bf16*)alloc((size_t)32 * 128 * 2);
  __bf16* wb_pT  = (__bf16*)alloc((size_t)5 * 128 * 128 * 2);

  __bf16* xe     = (__bf16*)alloc((size_t)E * 256 * 2);
  float*  msg_m  = (float*) alloc((size_t)E * 6400 * 4);
  float*  hbuf   = (float*) alloc((size_t)E * 64 * 4);
  __bf16* hb     = (__bf16*)alloc((size_t)E * 64 * 2);
  float*  radb   = (float*) alloc((size_t)E * 3840 * 4);
  __bf16* scaled = (__bf16*)alloc((size_t)E * 6400 * 2);
  float*  fc0o   = (float*) alloc((size_t)E * 896 * 4);
  float*  y1[4];
  for (int i = 0; i < 4; ++i) y1[i] = (float*)alloc((size_t)2 * E * (2 * 64 * SZ[i]) * 4);
  float*  msg1   = (float*) alloc((size_t)E * 1600 * 4);
  __bf16* gridT  = (__bf16*)alloc((size_t)E * 64 * 128 * 2);
  float*  s2b    = (float*) alloc((size_t)E * 1600 * 4);
  __bf16* sc2    = (__bf16*)alloc((size_t)E * 1600 * 2);
  float*  fc0o2  = (float*) alloc((size_t)E * 640 * 4);
  float*  y2[4];
  for (int i = 0; i < 4; ++i) y2[i] = (float*)alloc((size_t)2 * E * (2 * 128 * SZ[i]) * 4);
  float*  msg2m  = (float*) alloc((size_t)E * 3200 * 4);
  float*  logits = (float*) alloc((size_t)E * 8 * 4);
  float*  aex    = (float*) alloc((size_t)E * 8 * 4);
  float*  mxb    = (float*) alloc((size_t)NN * 8 * 4);
  float*  ssum   = (float*) alloc((size_t)NN * 8 * 4);
  float*  node   = (float*) alloc((size_t)NN * 3200 * 4);
  __bf16* nodeb  = (__bf16*)alloc((size_t)NN * 3200 * 2);
  (void)alloc((size_t)1 << 20);   // tail pad: async A-tile DMA may overshoot M

  auto cvt = [&](const float* in, __bf16* o, int Rin, int Cin, int Rout, int Cout, int tr) {
    long n = (long)Rout * Cout;
    convert_pad<<<(unsigned)((n + 255) / 256), 256, 0, stream>>>(in, o, Rin, Cin, Rout, Cout, tr);
  };
  auto gemm = [&](const __bf16* A, const __bf16* BT, const float* bias, float* C,
                  int M, int N, int K, int aR, long aGS, long aRS, long aBase,
                  int cR, long cGS, long cRS, long cBase) {
    dim3 g((unsigned)(N / 64), (unsigned)((M + 127) / 128));
    gemm_bf16<<<g, 256, 0, stream>>>(A, BT, bias, C, M, N, K, aR, aGS, aRS, aBase, cR, cGS, cRS, cBase);
  };

  // ---- weight conversion: transpose to [N][K] for B operands ----
  cvt(rw1, wb_rw1, 256, 64, 64, 256, 1);
  cvt(rw2, wb_rw2, 64, 64, 64, 64, 1);
  cvt(rw3, wb_rw3, 64, 3840, 3840, 64, 1);
  cvt(c1fw, wb_c1f, 1280, 896, 896, 1280, 1);
  for (int i = 0; i < 4; ++i)
    cvt(c1wm[i], wb_c1w[i], SZ[i] * 256, 2 * 64 * SZ[i], 2 * 64 * SZ[i], SZ[i] * 256, 1);
  cvt(c2fw, wb_c2f, 320, 640, 640, 320, 1);
  for (int i = 0; i < 4; ++i)
    cvt(c2wm[i], wb_c2w[i], SZ[i] * 64, 2 * 128 * SZ[i], 2 * 128 * SZ[i], SZ[i] * 64, 1);
  // proj_w[l] is [out][in] = already [N][K]
  for (int l = 0; l < 5; ++l)
    cvt(pw + (size_t)l * 128 * 128, wb_pT + (size_t)l * 128 * 128, 128, 128, 128, 128, 0);
  // A operands of the grid einsums: fragment-swizzled global layouts
  {
    long n = (long)112 * 32;
    convert_aswz<<<(unsigned)((n + 255) / 256), 256, 0, stream>>>(tgw, wb_tg, 100, 25, 112, 32, 0);
    n = (long)32 * 128;
    convert_aswz<<<(unsigned)((n + 255) / 256), 256, 0, stream>>>(fgw, wb_fgT, 100, 25, 32, 128, 1);
  }

  // ---- edge features + forward Wigner rotation (batched WMMA) ----
  {
    long n = (long)E * 256;
    xedge_kernel<<<(unsigned)((n + 255) / 256), 256, 0, stream>>>(edist, stab, ttab, an, ei, xe, E);
  }
  wigner_fwd<<<E, 256, 0, stream>>>(wig, x, ei, msg_m, E);

  // ---- radial MLP: 256->64 (LN+SiLU) -> 64 (LN+SiLU) -> 3840 ----
  gemm(xe, wb_rw1, rb1, hbuf, E, 64, 256, 1, 256, 0, 0, 1, 64, 0, 0);
  ln_silu_kernel<<<(E + 255) / 256, 256, 0, stream>>>(hbuf, rg1, rbe1, hb, E);
  gemm(hb, wb_rw2, rb2, hbuf, E, 64, 64, 1, 64, 0, 0, 1, 64, 0, 0);
  ln_silu_kernel<<<(E + 255) / 256, 256, 0, stream>>>(hbuf, rg2, rbe2, hb, E);
  gemm(hb, wb_rw3, rb3, radb, E, 3840, 64, 1, 64, 0, 0, 1, 3840, 0, 0);

  // ---- radial scaling of rotated messages (fp32 * fp32 -> bf16) ----
  {
    long n = (long)E * 6400;
    rad_scale<<<(unsigned)((n + 255) / 256), 256, 0, stream>>>(msg_m, radb, scaled, n);
  }

  // ---- SO(2) conv 1 ----
  gemm(scaled, wb_c1f, c1fb, fc0o, E, 896, 1280, 1, 6400, 0, 0, 1, 896, 0, 0);
  for (int i = 0; i < 4; ++i)
    gemm(scaled, wb_c1w[i], nullptr, y1[i], 2 * E, 2 * 64 * SZ[i], SZ[i] * 256,
         2, 6400, (long)SZ[i] * 256, (long)OFF[i] * 256, 1, 2 * 64 * SZ[i], 0, 0);
  {
    long n = (long)E * 1600;
    combine_kernel<<<(unsigned)((n + 255) / 256), 256, 0, stream>>>(
        fc0o, 896, 576, y1[0], y1[1], y1[2], y1[3], msg1, 64, n);
  }

  // ---- attention logits + segment softmax over tgt ----
  {
    long n = (long)E * 8;
    alpha_logits_kernel<<<(unsigned)((n + 255) / 256), 256, 0, stream>>>(fc0o, alg, albp, adot, logits, n);
    init_f<<<(unsigned)(((long)NN * 8 + 255) / 256), 256, 0, stream>>>(mxb, -3.0e38f, (long)NN * 8);
    init_f<<<(unsigned)(((long)NN * 8 + 255) / 256), 256, 0, stream>>>(ssum, 0.f, (long)NN * 8);
    smax_max<<<(unsigned)((n + 255) / 256), 256, 0, stream>>>(logits, ei, mxb, E);
    smax_sum<<<(unsigned)((n + 255) / 256), 256, 0, stream>>>(logits, ei, mxb, aex, ssum, E);
    smax_norm<<<(unsigned)((n + 255) / 256), 256, 0, stream>>>(ei, ssum, aex, E);
  }

  // ---- S2 grid activation (batched WMMA per edge) ----
  grid_fwd<<<E, 256, 0, stream>>>(wb_tg, msg1, gridT, E);
  grid_bwd<<<E, 256, 0, stream>>>(wb_fgT, gridT, s2b, E);
  {
    long n = (long)E * 1600;
    scaled2_kernel<<<(unsigned)((n + 255) / 256), 256, 0, stream>>>(fc0o, s2b, sc2, n);
  }

  // ---- SO(2) conv 2 ----
  gemm(sc2, wb_c2f, c2fb, fc0o2, E, 640, 320, 1, 1600, 0, 0, 1, 640, 0, 0);
  for (int i = 0; i < 4; ++i)
    gemm(sc2, wb_c2w[i], nullptr, y2[i], 2 * E, 2 * 128 * SZ[i], SZ[i] * 64,
         2, 1600, (long)SZ[i] * 64, (long)OFF[i] * 64, 1, 2 * 128 * SZ[i], 0, 0);
  {
    long n = (long)E * 3200;
    combine_kernel<<<(unsigned)((n + 255) / 256), 256, 0, stream>>>(
        fc0o2, 640, 0, y2[0], y2[1], y2[2], y2[3], msg2m, 128, n);
  }

  // ---- inverse Wigner + attention weighting + scatter to nodes ----
  init_f<<<(unsigned)(((long)NN * 3200 + 255) / 256), 256, 0, stream>>>(node, 0.f, (long)NN * 3200);
  wigner_bwd_attn<<<E, 256, 0, stream>>>(wig, msg2m, aex, ei, node, E);

  // ---- output projection per degree l, straight into d_out ----
  cvt(node, nodeb, NN * 25, 128, NN * 25, 128, 0);
  for (int l = 0; l < 5; ++l) {
    int R = 2 * l + 1;
    gemm(nodeb, wb_pT + (size_t)l * 128 * 128, nullptr, out,
         NN * R, 128, 128, R, 3200, 128, (long)l * l * 128,
         R, 3200, 128, (long)l * l * 128);
  }
  bias_row0<<<(unsigned)(((long)NN * 128 + 255) / 256), 256, 0, stream>>>(out, pb, NN);
}
